// ClusteringLayer_44873818308675
// MI455X (gfx1250) — matrix operand order repaired and checked by reading, not compiled, not run on hardware
//
#include <hip/hip_runtime.h>

// Student-t soft assignment (DEC clustering), ALPHA = 1.
// N = 262144, D = 64, K = 512, fp32 in/out. Output (512 MB) sets a ~22 us
// floor @ 23.3 TB/s -> minimize instructions per output element.
// Cross term via split-bf16 WMMA (hi*hi + lo*hi + hi*lo) on
// v_wmma_f32_16x16x32_bf16 ~ fp32 accuracy.
//
// Pre-pass converts centroids ONCE into d_ws as per-lane WMMA B fragments
// (bf16 hi/lo) + ||c||^2. A tile converted once per block (2 waves), staged
// via LDS. 32 rows x 512 cols per block; row-normalize fused.
// v3: hardware rcp + 1 Newton step (vs IEEE div expansion), direct LDS
// atomic row sums (no shuffle butterfly), immediate-offset stores.

typedef __attribute__((ext_vector_type(16))) __bf16 v16bf;
typedef __attribute__((ext_vector_type(8)))  float  v8f;
typedef __attribute__((ext_vector_type(4)))  float  v4f;

static constexpr int kD     = 64;
static constexpr int kK     = 512;
static constexpr int kTiles = kK / 16;                        // 32 col tiles
static constexpr int kFragCount = kTiles * 2 * 2 * 32;        // 4096 fragments
static constexpr size_t kFragBytes = (size_t)kFragCount * 32; // 128 KB
static constexpr size_t kCsqOff    = kFragBytes;              // +2 KB csq

#define WMMA_BF16(A, B, C) \
    __builtin_amdgcn_wmma_f32_16x16x32_bf16(false, (A), false, (B), (short)0, (C), false, false)

// rcp + one Newton-Raphson step: ~fp32-exact reciprocal in 3 instructions
__device__ __forceinline__ float fast_rcp(float d) {
    const float r = __builtin_amdgcn_rcpf(d);
    return r * fmaf(-d, r, 2.0f);
}

// ---- pre-pass: centroids -> split-bf16 B fragments + ||c||^2 in d_ws ----
// frag[gid], gid = t*128 + ks*64 + h*32 + lane  == [t][ks][h][lane] v16bf
// B frag K order used by this code (32x16 bf16): lane<16 col=lane, elem e -> K=e; lane>=16 K=16+e.
__global__ __launch_bounds__(256)
void centroid_prep_kernel(const float* __restrict__ cent, void* __restrict__ ws)
{
    const int gid = blockIdx.x * 256 + threadIdx.x;
    v16bf* frag = (v16bf*)ws;
    float* csq  = (float*)((char*)ws + kCsqOff);
    if (gid < kFragCount) {
        const int lane = gid & 31;
        const int h    = (gid >> 5) & 1;
        const int ks   = (gid >> 6) & 1;
        const int t    = gid >> 7;
        const int col  = t * 16 + (lane & 15);
        const int hi   = lane >> 4;
        const float* src = cent + (size_t)col * kD + 16 * hi + 32 * ks;
        v16bf o;
        #pragma unroll
        for (int e = 0; e < 16; ++e) {
            const float v = src[e];
            const __bf16 bh = (__bf16)v;
            o[e] = h ? (__bf16)(v - (float)bh) : bh;
        }
        frag[gid] = o;
    } else if (gid < kFragCount + kK) {
        const int col = gid - kFragCount;
        const float* src = cent + (size_t)col * kD;
        float s = 0.0f;
        #pragma unroll
        for (int d = 0; d < kD; ++d) s += src[d] * src[d];
        csq[col] = s;
    }
}

// ---- main kernel: 256 threads = 8 waves; 32 rows x 512 cols per block ----
__global__ __launch_bounds__(256)
void clustering_q_kernel(const float* __restrict__ x,
                         const void* __restrict__ ws,
                         float* __restrict__ out)
{
    const int tid  = threadIdx.x;
    const int lane = tid & 31;
    const int wave = tid >> 5;
    const int hi   = (lane >> 4) & 1;
    const int lme  = lane & 15;
    const int row_base = blockIdx.x * 32;

    __shared__ v16bf s_afrag[2][2][2][32];   // [rt][ks][hi/lo][lane], 8 KB
    __shared__ float s_xsq[32];              // 1 + ||x||^2 per row
    __shared__ float s_rowsum[32];

    if (tid < 32) s_rowsum[tid] = 0.0f;

    // waves 0,1 each convert one 16-row A tile into LDS fragments (once/block)
    // A frag K order used by this code (16x32 bf16): lane<16 row=lane, e -> K=e;
    // lane>=16 same row, K=16+e.
    if (wave < 2) {
        const int rt = wave;
        const float* arow = x + (size_t)(row_base + rt * 16 + lme) * kD;
        float xsq = 0.0f;
        #pragma unroll
        for (int ks = 0; ks < 2; ++ks) {
            v16bf h, l;
            #pragma unroll
            for (int e = 0; e < 16; ++e) {
                const int dim = e + 16 * hi + 32 * ks;
                const float v = arow[dim];
                xsq += v * v;
                const __bf16 bh = (__bf16)v;
                h[e] = bh;
                l[e] = (__bf16)(v - (float)bh);
            }
            s_afrag[rt][ks][0][lane] = h;
            s_afrag[rt][ks][1][lane] = l;
        }
        xsq += __shfl_xor(xsq, 16, 32);      // combine complementary dim halves
        if (lane < 16) s_xsq[rt * 16 + lane] = xsq + 1.0f;   // fold the +1
    }
    __syncthreads();

    v16bf a_hi[2][2], a_lo[2][2];
    #pragma unroll
    for (int rt = 0; rt < 2; ++rt)
        #pragma unroll
        for (int ks = 0; ks < 2; ++ks) {
            a_hi[rt][ks] = s_afrag[rt][ks][0][lane];
            a_lo[rt][ks] = s_afrag[rt][ks][1][lane];
        }

    // 1+||x||^2 for the 8 rows of each C fragment half (vectorized LDS reads)
    float xsq1[2][8];
    #pragma unroll
    for (int rt = 0; rt < 2; ++rt) {
        const v4f* p = (const v4f*)&s_xsq[rt * 16 + 8 * hi];
        const v4f p0 = p[0], p1 = p[1];
        #pragma unroll
        for (int j = 0; j < 4; ++j) { xsq1[rt][j] = p0[j]; xsq1[rt][4 + j] = p1[j]; }
    }

    // ---- B fragments from ws (coalesced, L2-resident) + split WMMA ----
    // per-wave bases so every load below is base + constant immediate offset
    const v16bf* bw   = (const v16bf*)ws + wave * 512 + lane;          // 4 tiles
    const float* csqw = (const float*)((const char*)ws + kCsqOff) + wave * 64 + lme;

    v8f   acc[2][4];
    float csq[4];
    #pragma unroll
    for (int tt = 0; tt < 4; ++tt) {
        const v16bf bh0 = bw[tt * 128 +  0];
        const v16bf bl0 = bw[tt * 128 + 32];
        const v16bf bh1 = bw[tt * 128 + 64];
        const v16bf bl1 = bw[tt * 128 + 96];
        csq[tt] = csqw[tt * 16];
        #pragma unroll
        for (int rt = 0; rt < 2; ++rt) {
            v8f c = {};
            c = WMMA_BF16(a_hi[rt][0], bh0, c);
            c = WMMA_BF16(a_lo[rt][0], bh0, c);
            c = WMMA_BF16(a_hi[rt][0], bl0, c);
            c = WMMA_BF16(a_hi[rt][1], bh1, c);
            c = WMMA_BF16(a_lo[rt][1], bh1, c);
            c = WMMA_BF16(a_hi[rt][1], bl1, c);
            acc[rt][tt] = c;
        }
    }

    // ---- epilogue: q = 1/(1+d2); per-lane 4-col partials -> LDS atomics ----
    // C/D layout: elem r -> row = r + 8*hi (+16*rt), col = lme + 16*tile
    #pragma unroll
    for (int rt = 0; rt < 2; ++rt) {
        float rs[8];
        #pragma unroll
        for (int r = 0; r < 8; ++r) rs[r] = 0.0f;
        #pragma unroll
        for (int tt = 0; tt < 4; ++tt) {
            #pragma unroll
            for (int r = 0; r < 8; ++r) {
                const float d = fmaf(acc[rt][tt][r], -2.0f, xsq1[rt][r]) + csq[tt];
                const float q = fast_rcp(d);         // d == 1 + dist2
                acc[rt][tt][r] = q;
                rs[r] += q;
            }
        }
        #pragma unroll
        for (int r = 0; r < 8; ++r)                  // ds_add_f32, no shuffles
            atomicAdd(&s_rowsum[rt * 16 + 8 * hi + r], rs[r]);
    }
    __syncthreads();

    // ---- normalize + store: single vaddr, all-immediate offsets ----
    float* op = out + (size_t)(row_base + 8 * hi) * kK + wave * 64 + lme;
    #pragma unroll
    for (int rt = 0; rt < 2; ++rt) {
        float inv[8];
        const v4f* p = (const v4f*)&s_rowsum[rt * 16 + 8 * hi];
        const v4f p0 = p[0], p1 = p[1];
        #pragma unroll
        for (int j = 0; j < 4; ++j) { inv[j] = fast_rcp(p0[j]); inv[4 + j] = fast_rcp(p1[j]); }
        #pragma unroll
        for (int tt = 0; tt < 4; ++tt) {
            #pragma unroll
            for (int r = 0; r < 8; ++r)
                op[(size_t)(rt * 16 + r) * kK + tt * 16] = acc[rt][tt][r] * inv[r];
        }
    }
}

extern "C" void kernel_launch(void* const* d_in, const int* in_sizes, int n_in,
                              void* d_out, int out_size, void* d_ws, size_t ws_size,
                              hipStream_t stream) {
    (void)n_in; (void)out_size; (void)ws_size;
    const float* x    = (const float*)d_in[0];   // [N, 64]
    const float* cent = (const float*)d_in[1];   // [512, 64]
    float* out        = (float*)d_out;           // [N, 512]
    const int n_rows  = in_sizes[0] / kD;        // 262144

    const int prep_threads = kFragCount + kK;    // 4608
    centroid_prep_kernel<<<(prep_threads + 255) / 256, 256, 0, stream>>>(cent, d_ws);
    clustering_q_kernel<<<n_rows / 32, 256, 0, stream>>>(x, d_ws, out);
}
